// Pointnet2Backbone_73796128080209
// MI455X (gfx1250) — matrix operand order repaired and checked by reading, not compiled, not run on hardware
//
#include <hip/hip_runtime.h>
#include <hip/hip_bf16.h>

// ---------------------------------------------------------------------------
// PointNet++ backbone for MI455X (gfx1250).
// MLP layers = batched GEMMs via V_WMMA_F32_16X16X32_F16 with direct
// contiguous fragment loads (no LDS), software-pipelined K loop (double-
// buffered fragments), and 16x32 output tile per wave (A-fragment reuse
// across two accumulators).
// ---------------------------------------------------------------------------

typedef __attribute__((ext_vector_type(16))) _Float16 v16h;
typedef __attribute__((ext_vector_type(8)))  _Float16 v8h;
typedef __attribute__((ext_vector_type(8)))  float    v8f;

#define EPS_BN 1e-5f

// ---------------------------------------------------------------------------
// Farthest point sampling: one block per batch, sequential argmax steps.
// ---------------------------------------------------------------------------
__global__ void fps_kernel(const float* __restrict__ xyz, float* __restrict__ mind,
                           int* __restrict__ idx, int N, int npoint)
{
    const int b   = blockIdx.x;
    const int tid = threadIdx.x;
    const int T   = blockDim.x;
    const float* P = xyz + (long long)b * N * 3;
    float* md = mind + (long long)b * N;

    for (int i = tid; i < N; i += T) md[i] = 1e10f;

    __shared__ float s_best[256];
    __shared__ int   s_bidx[256];
    __shared__ int   s_cur;
    if (tid == 0) { s_cur = 0; idx[b * npoint] = 0; }
    __syncthreads();

    for (int it = 1; it < npoint; ++it) {
        const int cur = s_cur;
        const float cx = P[cur * 3 + 0];
        const float cy = P[cur * 3 + 1];
        const float cz = P[cur * 3 + 2];
        float best = -1.0f; int bi = 0;
        for (int i = tid; i < N; i += T) {
            float dx = P[i * 3 + 0] - cx;
            float dy = P[i * 3 + 1] - cy;
            float dz = P[i * 3 + 2] - cz;
            float d  = dx * dx + dy * dy + dz * dz;
            float m  = md[i];
            if (d < m) { m = d; md[i] = m; }
            if (m > best) { best = m; bi = i; }
        }
        s_best[tid] = best; s_bidx[tid] = bi;
        __syncthreads();
        for (int off = T >> 1; off > 0; off >>= 1) {
            if (tid < off) {
                if (s_best[tid + off] > s_best[tid] ||
                    (s_best[tid + off] == s_best[tid] && s_bidx[tid + off] < s_bidx[tid])) {
                    s_best[tid] = s_best[tid + off];
                    s_bidx[tid] = s_bidx[tid + off];
                }
            }
            __syncthreads();
        }
        if (tid == 0) { s_cur = s_bidx[0]; idx[b * npoint + it] = s_bidx[0]; }
        __syncthreads();
    }
}

__global__ void gather_xyz_kernel(const float* __restrict__ xyz, const int* __restrict__ idx,
                                  float* __restrict__ out, int N, int npoint, int total)
{
    int t = blockIdx.x * blockDim.x + threadIdx.x;
    if (t >= total) return;
    int b = t / npoint;
    int j = idx[t];
    const float* p = xyz + ((long long)b * N + j) * 3;
    out[t * 3 + 0] = p[0];
    out[t * 3 + 1] = p[1];
    out[t * 3 + 2] = p[2];
}

// First nsample in-radius indices in ascending order, padded with first hit.
__global__ void ball_query_kernel(const float* __restrict__ xyz, const float* __restrict__ new_xyz,
                                  int* __restrict__ nidx, int N, int npoint, int nsample,
                                  float r2, int total)
{
    int t = blockIdx.x * blockDim.x + threadIdx.x;
    if (t >= total) return;
    int b = t / npoint;
    const float* P = xyz + (long long)b * N * 3;
    float cx = new_xyz[t * 3 + 0], cy = new_xyz[t * 3 + 1], cz = new_xyz[t * 3 + 2];
    int* out = nidx + (long long)t * nsample;
    int cnt = 0, first = 0;
    for (int i = 0; i < N && cnt < nsample; ++i) {
        float dx = P[i * 3 + 0] - cx;
        float dy = P[i * 3 + 1] - cy;
        float dz = P[i * 3 + 2] - cz;
        if (dx * dx + dy * dy + dz * dz < r2) {
            if (cnt == 0) first = i;
            out[cnt++] = i;
        }
    }
    for (; cnt < nsample; ++cnt) out[cnt] = first;
}

// Build grouped GEMM rows: [xyz - center, (feats)] zero-padded to Kpad.
__global__ void group_kernel(const float* __restrict__ xyz, const float* __restrict__ new_xyz,
                             const int* __restrict__ nidx, const _Float16* __restrict__ feats,
                             int Cf, int Fstride, int Nin, _Float16* __restrict__ X,
                             int Kpad, int npoint, int nsample, int total)
{
    int t = blockIdx.x * blockDim.x + threadIdx.x;
    if (t >= total) return;
    int c = (t / nsample) % npoint;
    int b = t / (nsample * npoint);
    int j = nidx[t];
    const float* p = xyz + ((long long)b * Nin + j) * 3;
    const float* q = new_xyz + ((long long)b * npoint + c) * 3;
    _Float16* row = X + (long long)t * Kpad;
    row[0] = (_Float16)(p[0] - q[0]);
    row[1] = (_Float16)(p[1] - q[1]);
    row[2] = (_Float16)(p[2] - q[2]);
    int k = 3;
    if (feats) {
        const _Float16* f = feats + ((long long)b * Nin + j) * Fstride;
        for (int c2 = 0; c2 < Cf; ++c2) row[k++] = f[c2];
    }
    for (; k < Kpad; ++k) row[k] = (_Float16)0.0f;
}

// Group-all concat for SA3: [xyz, feats] per point, zero-padded to Kpad.
__global__ void concat_all_kernel(const float* __restrict__ xyz, const _Float16* __restrict__ feats,
                                  int Cf, int Fstride, _Float16* __restrict__ X, int Kpad, int rows)
{
    int t = blockIdx.x * blockDim.x + threadIdx.x;
    if (t >= rows) return;
    _Float16* row = X + (long long)t * Kpad;
    row[0] = (_Float16)xyz[t * 3 + 0];
    row[1] = (_Float16)xyz[t * 3 + 1];
    row[2] = (_Float16)xyz[t * 3 + 2];
    const _Float16* f = feats + (long long)t * Fstride;
    int k = 3;
    for (int c2 = 0; c2 < Cf; ++c2) row[k++] = f[c2];
    for (; k < Kpad; ++k) row[k] = (_Float16)0.0f;
}

// Convert f32 weights (Nact,Kact) -> zero-padded f16 (Npad,Kpad).
__global__ void wpad_kernel(const float* __restrict__ W, int Nact, int Kact,
                            _Float16* __restrict__ Wp, int Npad, int Kpad)
{
    int t = blockIdx.x * blockDim.x + threadIdx.x;
    if (t >= Npad * Kpad) return;
    int n = t / Kpad, k = t % Kpad;
    float v = (n < Nact && k < Kact) ? W[(long long)n * Kact + k] : 0.0f;
    Wp[t] = (_Float16)v;
}

// ---------------------------------------------------------------------------
// WMMA GEMM, direct fragment loads, double-buffered K pipeline, 16x32 tile.
//   X  : (M, Kpad) f16 row-major, zero-padded, M % 16 == 0
//   Wp : (Npad, Kpad) f16 row-major, zero-padded, Npad % 32 == 0
//   Y = act( X * Wp^T [+BN | +bias] ); f16 out (cols >= Nact forced to 0)
//   and/or f32 out (cols < Nact only).
// ---------------------------------------------------------------------------
__global__ void gemm_wmma_direct(const _Float16* __restrict__ X, int Kpad,
                                 const _Float16* __restrict__ Wp,
                                 const float* __restrict__ g, const float* __restrict__ bb,
                                 const float* __restrict__ mm, const float* __restrict__ vv,
                                 const float* __restrict__ bias, int relu, int Nact,
                                 _Float16* __restrict__ Y16, int Ystride,
                                 float* __restrict__ Y32, int Y32stride,
                                 int nt2_count, long long total_waves)
{
    const long long w = (long long)blockIdx.x * (blockDim.x >> 5) + (threadIdx.x >> 5);
    if (w >= total_waves) return;
    const int lane = threadIdx.x & 31;
    const int mt  = (int)(w / nt2_count);
    const int nt2 = (int)(w % nt2_count);

    const int rl = lane & 15;      // A row / B col within tile
    const int hi = lane >> 4;      // 0 or 1 (upper half-wave)

    // A frag: lane holds row (mt*16+rl), halves [kb..kb+7] and [kb+16..kb+23], kb = 8*hi
    const v8h* Ap  = (const v8h*)(X + (long long)(mt * 16 + rl) * Kpad + hi * 8);
    // B frags: lane holds col (nt2*32+rl) and (+16), halves [kb..kb+7] and [kb+16..kb+23], kb = 8*hi
    const v8h* Bp0 = (const v8h*)(Wp + (long long)(nt2 * 32 + rl) * Kpad + hi * 8);
    const v8h* Bp1 = Bp0 + 2 * Kpad / 8 * 8 / 8;  // +16 rows = 16*Kpad halves
    Bp1 = (const v8h*)((const _Float16*)Bp0 + (long long)16 * Kpad);
    const int step = Kpad >> 3;    // Kpad halves (one X/Wp row) in v8h units

    auto ldA = [](const v8h* P) -> v16h {
        v8h lo = P[0], hh = P[2];  // K runs {kb..kb+7} and {kb+16..kb+23}
        return __builtin_shufflevector(lo, hh, 0,1,2,3,4,5,6,7,8,9,10,11,12,13,14,15);
    };
    auto ldB = [](const v8h* P) -> v16h {
        v8h lo = P[0], hh = P[2];  // K runs {kb..kb+7} and {kb+16..kb+23}
        return __builtin_shufflevector(lo, hh, 0,1,2,3,4,5,6,7,8,9,10,11,12,13,14,15);
    };

    v16h aC  = ldA(Ap);
    v16h b0C = ldB(Bp0);
    v16h b1C = ldB(Bp1);
    v8f acc0 = {}, acc1 = {};

    for (int kk = 0; kk < Kpad; kk += 32) {
        Ap += step; Bp0 += step; Bp1 += step;
        v16h aN, b0N, b1N;
        const bool more = (kk + 32 < Kpad);
        if (more) {
            aN  = ldA(Ap);                         // issue next-K loads before
            b0N = ldB(Bp0);                        // consuming current frags
            b1N = ldB(Bp1);
            __builtin_prefetch((const void*)(Ap + step), 0, 1);  // global_prefetch_b8
        }
        acc0 = __builtin_amdgcn_wmma_f32_16x16x32_f16(
            false, aC, false, b0C, (short)0, acc0, false, false);
        acc1 = __builtin_amdgcn_wmma_f32_16x16x32_f16(
            false, aC, false, b1C, (short)0, acc1, false, false);
        if (more) { aC = aN; b0C = b0N; b1C = b1N; }
    }

    // D layout: VGPR r -> row (r + 8*hi), col = lane&15 within each tile.
    const int mbase = mt * 16 + hi * 8;
    const int ncol0 = nt2 * 32 + rl;

    #pragma unroll
    for (int half = 0; half < 2; ++half) {
        const int n = ncol0 + half * 16;
        const v8f& acc = half ? acc1 : acc0;
        const bool inN = (n < Nact);
        float s = 1.0f, t = 0.0f;
        if (inN) {
            if (g) {
                const float sv = g[n] * rsqrtf(vv[n] + EPS_BN);
                s = sv;
                t = bb[n] - mm[n] * sv;
            } else if (bias) {
                t = bias[n];
            }
        }
        #pragma unroll
        for (int r = 0; r < 8; ++r) {
            const int m = mbase + r;
            float y = inN ? (acc[r] * s + t) : 0.0f;
            if (relu && y < 0.0f) y = 0.0f;
            if (Y16) Y16[(long long)m * Ystride + n] = (_Float16)y;
            if (Y32 && inN) Y32[(long long)m * Y32stride + n] = y;
        }
    }
}

// Max over nsample rows per group; optional f16 and f32 outputs.
__global__ void maxpool_kernel(const _Float16* __restrict__ Yin, int stride, int nsample, int C,
                               _Float16* __restrict__ out16, int ostride,
                               float* __restrict__ out32, int groups)
{
    int t = blockIdx.x * blockDim.x + threadIdx.x;
    if (t >= groups * C) return;
    int gr = t / C, c = t % C;
    const _Float16* p = Yin + (long long)gr * nsample * stride + c;
    float mx = -1e30f;
    for (int s = 0; s < nsample; ++s) {
        float v = (float)p[(long long)s * stride];
        if (v > mx) mx = v;
    }
    if (out16) out16[(long long)gr * ostride + c] = (_Float16)mx;
    if (out32) out32[(long long)gr * C + c] = mx;
}

// ---------------------------------------------------------------------------
extern "C" void kernel_launch(void* const* d_in, const int* in_sizes, int n_in,
                              void* d_out, int out_size, void* d_ws, size_t ws_size,
                              hipStream_t stream)
{
    (void)in_sizes; (void)n_in; (void)out_size; (void)ws_size;

    const int B = 16, N1 = 16384;
    const int NP1 = 128, NS1 = 64;   const float R1SQ = 0.2f * 0.2f;
    const int NP2 = 64,  NS2 = 16;   const float R2SQ = 0.4f * 0.4f;

    // ---- unpack inputs (pytree flatten order) ----
    struct Layer { const float *W, *g, *b, *m, *v; };
    int p = 0;
    const float* pc = (const float*)d_in[p++];
    auto rd5 = [&](Layer& L) {
        L.W = (const float*)d_in[p++]; L.g = (const float*)d_in[p++];
        L.b = (const float*)d_in[p++]; L.m = (const float*)d_in[p++];
        L.v = (const float*)d_in[p++];
    };
    Layer sa1[3], sa2[3], sa3[3], fcl1, fcl2;
    for (int i = 0; i < 3; ++i) rd5(sa1[i]);
    for (int i = 0; i < 3; ++i) rd5(sa2[i]);
    for (int i = 0; i < 3; ++i) rd5(sa3[i]);
    rd5(fcl1); rd5(fcl2);
    const float* fcW3 = (const float*)d_in[p++];
    const float* fcb3 = (const float*)d_in[p++];

    float* out_f = (float*)d_out;  // feats (16,1024) then logits (16,18)

    // ---- workspace carve ----
    char* ws = (char*)d_ws;
    size_t off = 0;
    auto alloc = [&](size_t bytes) -> void* {
        void* r = ws + off;
        off += (bytes + 255) & ~(size_t)255;
        return r;
    };
    auto cdiv = [](long long a, long long b) { return (int)((a + b - 1) / b); };

    float* mind  = (float*)alloc(sizeof(float) * B * N1);
    int*   idx1  = (int*)alloc(sizeof(int) * B * NP1);
    float* nx1   = (float*)alloc(sizeof(float) * B * NP1 * 3);
    int*   nidx1 = (int*)alloc(sizeof(int) * B * NP1 * NS1);
    int*   idx2  = (int*)alloc(sizeof(int) * B * NP2);
    float* nx2   = (float*)alloc(sizeof(float) * B * NP2 * 3);
    int*   nidx2 = (int*)alloc(sizeof(int) * B * NP2 * NS2);
    const long long M1 = (long long)B * NP1 * NS1;   // 131072
    const long long M2 = (long long)B * NP2 * NS2;   // 16384
    const long long M3 = (long long)B * NP2;         // 1024
    _Float16* actA    = (_Float16*)alloc(sizeof(_Float16) * M1 * 96);
    _Float16* actB    = (_Float16*)alloc(sizeof(_Float16) * M1 * 96);
    _Float16* feats1  = (_Float16*)alloc(sizeof(_Float16) * B * NP1 * 64);
    _Float16* feats2  = (_Float16*)alloc(sizeof(_Float16) * B * NP2 * 128);
    _Float16* feats3h = (_Float16*)alloc(sizeof(_Float16) * B * 1024);

    // ---- pre-pad weights to f16 (Npad,Kpad) ----
    auto padW = [&](const float* W, int Nact, int Kact, int Npad, int Kpad) -> _Float16* {
        _Float16* Wp = (_Float16*)alloc(sizeof(_Float16) * (size_t)Npad * Kpad);
        wpad_kernel<<<cdiv((long long)Npad * Kpad, 256), 256, 0, stream>>>(W, Nact, Kact, Wp, Npad, Kpad);
        return Wp;
    };
    _Float16* w_sa1[3] = { padW(sa1[0].W, 32, 3, 32, 32),
                           padW(sa1[1].W, 32, 32, 32, 32),
                           padW(sa1[2].W, 64, 32, 64, 32) };
    _Float16* w_sa2[3] = { padW(sa2[0].W, 62, 67, 64, 96),
                           padW(sa2[1].W, 64, 62, 64, 64),
                           padW(sa2[2].W, 128, 64, 128, 64) };
    _Float16* w_sa3[3] = { padW(sa3[0].W, 128, 131, 128, 160),
                           padW(sa3[1].W, 512, 128, 512, 128),
                           padW(sa3[2].W, 1024, 512, 1024, 512) };
    _Float16* w_fc1 = padW(fcl1.W, 512, 1024, 512, 1024);
    _Float16* w_fc2 = padW(fcl2.W, 256, 512, 256, 512);
    _Float16* w_fc3 = padW(fcW3, 18, 256, 32, 256);

    // ---- GEMM launcher: one wave / 16x32 tile, 8 waves / block ----
    auto gemm = [&](const _Float16* X, long long M, int Kpad, const _Float16* Wp, int Npad,
                    int Nact, const Layer* bn, const float* bias, int relu,
                    _Float16* Y16, int Ystride, float* Y32, int Y32stride) {
        const int ntc2 = Npad / 32;
        const long long total = (M / 16) * ntc2;
        gemm_wmma_direct<<<cdiv(total, 8), 256, 0, stream>>>(
            X, Kpad, Wp,
            bn ? bn->g : nullptr, bn ? bn->b : nullptr,
            bn ? bn->m : nullptr, bn ? bn->v : nullptr,
            bias, relu, Nact, Y16, Ystride, Y32, Y32stride, ntc2, total);
    };

    // ======================= SA1 =======================
    fps_kernel<<<B, 256, 0, stream>>>(pc, mind, idx1, N1, NP1);
    gather_xyz_kernel<<<cdiv(B * NP1, 256), 256, 0, stream>>>(pc, idx1, nx1, N1, NP1, B * NP1);
    ball_query_kernel<<<cdiv(B * NP1, 256), 256, 0, stream>>>(pc, nx1, nidx1, N1, NP1, NS1, R1SQ, B * NP1);
    group_kernel<<<cdiv(M1, 256), 256, 0, stream>>>(pc, nx1, nidx1, nullptr, 0, 0, N1,
                                                    actA, 32, NP1, NS1, (int)M1);
    gemm(actA, M1, 32, w_sa1[0], 32, 32, &sa1[0], nullptr, 1, actB, 32, nullptr, 0);
    gemm(actB, M1, 32, w_sa1[1], 32, 32, &sa1[1], nullptr, 1, actA, 32, nullptr, 0);
    gemm(actA, M1, 32, w_sa1[2], 64, 64, &sa1[2], nullptr, 1, actB, 64, nullptr, 0);
    maxpool_kernel<<<cdiv((long long)B * NP1 * 64, 256), 256, 0, stream>>>(
        actB, 64, NS1, 64, feats1, 64, nullptr, B * NP1);

    // ======================= SA2 =======================
    fps_kernel<<<B, 256, 0, stream>>>(nx1, mind, idx2, NP1, NP2);
    gather_xyz_kernel<<<cdiv(B * NP2, 256), 256, 0, stream>>>(nx1, idx2, nx2, NP1, NP2, B * NP2);
    ball_query_kernel<<<cdiv(B * NP2, 256), 256, 0, stream>>>(nx1, nx2, nidx2, NP1, NP2, NS2, R2SQ, B * NP2);
    group_kernel<<<cdiv(M2, 256), 256, 0, stream>>>(nx1, nx2, nidx2, feats1, 64, 64, NP1,
                                                    actA, 96, NP2, NS2, (int)M2);
    gemm(actA, M2, 96, w_sa2[0], 64, 62, &sa2[0], nullptr, 1, actB, 64, nullptr, 0);
    gemm(actB, M2, 64, w_sa2[1], 64, 64, &sa2[1], nullptr, 1, actA, 64, nullptr, 0);
    gemm(actA, M2, 64, w_sa2[2], 128, 128, &sa2[2], nullptr, 1, actB, 128, nullptr, 0);
    maxpool_kernel<<<cdiv((long long)B * NP2 * 128, 256), 256, 0, stream>>>(
        actB, 128, NS2, 128, feats2, 128, nullptr, B * NP2);

    // ======================= SA3 (group-all) =======================
    concat_all_kernel<<<cdiv(M3, 256), 256, 0, stream>>>(nx2, feats2, 128, 128, actA, 160, (int)M3);
    gemm(actA, M3, 160, w_sa3[0], 128, 128, &sa3[0], nullptr, 1, actB, 128, nullptr, 0);
    gemm(actB, M3, 128, w_sa3[1], 512, 512, &sa3[1], nullptr, 1, actA, 512, nullptr, 0);
    gemm(actA, M3, 512, w_sa3[2], 1024, 1024, &sa3[2], nullptr, 1, actB, 1024, nullptr, 0);
    maxpool_kernel<<<cdiv((long long)B * 1024, 256), 256, 0, stream>>>(
        actB, 1024, NP2, 1024, feats3h, 1024, out_f, B);

    // ======================= FC head =======================
    gemm(feats3h, B, 1024, w_fc1, 512, 512, &fcl1, nullptr, 1, actA, 512, nullptr, 0);
    gemm(actA,    B, 512,  w_fc2, 256, 256, &fcl2, nullptr, 1, actB, 256, nullptr, 0);
    gemm(actB,    B, 256,  w_fc3, 32, 18, nullptr, fcb3, 0, nullptr, 0,
         out_f + (long long)B * 1024, 18);
}
